// GraphSAGEConv_60696477827758
// MI455X (gfx1250) — matrix operand order repaired and checked by reading, not compiled, not run on hardware
//
#include <hip/hip_runtime.h>
#include <hip/hip_bf16.h>

// GraphSAGE/GCN 3-layer forward for MI455X (gfx1250, wave32).
// Bottleneck: ~102M global_atomic_add_f32 per layer on an L2-resident (192MB)
// working set. GEMM (4.9 GFLOP total) runs through v_wmma_f32_16x16x32_f16,
// fully fused with combine + bias + LayerNorm + ELU to kill staging traffic.

#define LN_EPS 1e-5f

typedef _Float16 v8h  __attribute__((ext_vector_type(8)));
typedef _Float16 v16h __attribute__((ext_vector_type(16)));
typedef float    v8f  __attribute__((ext_vector_type(8)));

// ---------------------------------------------------------------- utilities
__global__ void k_zero_f32(float* __restrict__ p, long n) {
  long i = (long)blockIdx.x * blockDim.x + threadIdx.x;
  long stride = (long)gridDim.x * blockDim.x;
  for (; i < n; i += stride) p[i] = 0.0f;
}

__global__ void k_f32_to_f16(const float* __restrict__ in, _Float16* __restrict__ out, long n) {
  long i = (long)blockIdx.x * blockDim.x + threadIdx.x;
  long stride = (long)gridDim.x * blockDim.x;
  for (; i < n; i += stride) out[i] = (_Float16)in[i];
}

// ---------------------------------------------------------------- degrees
__global__ void k_degree(const int* __restrict__ dst, float* __restrict__ deg, int nE) {
  int e = blockIdx.x * blockDim.x + threadIdx.x;
  if (e < nE)
    __hip_atomic_fetch_add(&deg[dst[e]], 1.0f, __ATOMIC_RELAXED, __HIP_MEMORY_SCOPE_AGENT);
}

__global__ void k_inv(float* __restrict__ deg, int n) {
  int i = blockIdx.x * blockDim.x + threadIdx.x;
  if (i < n) deg[i] = 1.0f / (deg[i] + 1.0f);
}

// ---------------------------------------------------------------- scatter-add
// 32 lanes per edge; float4 gather from feats[src], 4x global_atomic_add_f32
// into neigh[dst]. Both arrays L2-resident.
__global__ void k_scatter(const float* __restrict__ feats, const int* __restrict__ src,
                          const int* __restrict__ dst, float* __restrict__ neigh, int nE) {
  long tid = (long)blockIdx.x * blockDim.x + threadIdx.x;
  int e = (int)(tid >> 5);
  if (e >= nE) return;
  int lane = (int)(tid & 31);
  const float4 v = *(const float4*)(feats + (long)src[e] * 128 + lane * 4);
  float* o = neigh + (long)dst[e] * 128 + lane * 4;
  __hip_atomic_fetch_add(o + 0, v.x, __ATOMIC_RELAXED, __HIP_MEMORY_SCOPE_AGENT);
  __hip_atomic_fetch_add(o + 1, v.y, __ATOMIC_RELAXED, __HIP_MEMORY_SCOPE_AGENT);
  __hip_atomic_fetch_add(o + 2, v.z, __ATOMIC_RELAXED, __HIP_MEMORY_SCOPE_AGENT);
  __hip_atomic_fetch_add(o + 3, v.w, __ATOMIC_RELAXED, __HIP_MEMORY_SCOPE_AGENT);
}

// ---------------------------------------------------------------- fused layer
// One wave owns a 16-row block and ALL 128 output columns:
//   h[r][k] = (neigh[r][k] + feats[r][k]) * inv[r]          (built once, f16)
//   acc[t]  = h(16x128) @ W[t-th 16 out-ch]^T               (4x wmma per tile)
//   y       = elu(layernorm(acc + bias) * gamma + beta)     (in-register LN)
//
// A fragment (16x32 f16, ISA 7.12.2): lanes 0-15 row=base+lane hold
//   K = k0+{0..7} (elems 0-7) and k0+{16..23} (elems 8-15); lanes 16-31 same
//   rows, K shifted by +8. B fragment (32x16): column = lane&15; same K
//   order as A (k0+8h+{0..7}, k0+16+8h+{0..7}). D (8 VGPRs): elem j -> row
//   half_sel*8+j, col = lane&15. So row m of the 16x128 output lives at
//   element j of all 8 accumulators across one 16-lane half -> LN reduction is
//   4 shfl_xor steps (masks 8,4,2,1).
//
// NOTE: safe for out == neigh: all reads of this wave's rows land in registers
// before any store, and waves never touch other row blocks.
__global__ void k_sage_layer(const float* __restrict__ feats,  // [nRows][128]
                             const float* __restrict__ neigh,  // [nRows][128]
                             const float* __restrict__ inv,    // [nRows]
                             const _Float16* __restrict__ Wh,  // [128 out][128 in]
                             const float* __restrict__ bl,     // [128]
                             const float* __restrict__ gm,     // [128]
                             const float* __restrict__ bt,     // [128]
                             float* __restrict__ out,          // [nRows][128]
                             int nRows) {
  const int wave = (int)(((long)blockIdx.x * blockDim.x + threadIdx.x) >> 5);
  const int lane = threadIdx.x & 31;
  const int rowBase = wave << 4;
  if (rowBase >= nRows) return;                 // wave-uniform exit

  const int half_sel = lane >> 4;               // 0: lanes 0-15, 1: lanes 16-31
  const int lane16 = lane & 15;
  int r = rowBase + lane16;
  if (r >= nRows) r = nRows - 1;                // clamp stays inside own block
  const float scale = inv[r];
  const float* __restrict__ frow = feats + (long)r * 128;
  const float* __restrict__ grow = neigh + (long)r * 128;

  // ---- build the four A fragments once (K = 0..127) ----
  v16h afrag[4];
#pragma unroll
  for (int kk = 0; kk < 4; ++kk) {
    const int ka = kk * 32 + half_sel * 8;
    const float4 f0 = *(const float4*)(frow + ka);
    const float4 f1 = *(const float4*)(frow + ka + 4);
    const float4 f2 = *(const float4*)(frow + ka + 16);
    const float4 f3 = *(const float4*)(frow + ka + 20);
    const float4 g0 = *(const float4*)(grow + ka);
    const float4 g1 = *(const float4*)(grow + ka + 4);
    const float4 g2 = *(const float4*)(grow + ka + 16);
    const float4 g3 = *(const float4*)(grow + ka + 20);
    afrag[kk][0]  = (_Float16)((f0.x + g0.x) * scale);
    afrag[kk][1]  = (_Float16)((f0.y + g0.y) * scale);
    afrag[kk][2]  = (_Float16)((f0.z + g0.z) * scale);
    afrag[kk][3]  = (_Float16)((f0.w + g0.w) * scale);
    afrag[kk][4]  = (_Float16)((f1.x + g1.x) * scale);
    afrag[kk][5]  = (_Float16)((f1.y + g1.y) * scale);
    afrag[kk][6]  = (_Float16)((f1.z + g1.z) * scale);
    afrag[kk][7]  = (_Float16)((f1.w + g1.w) * scale);
    afrag[kk][8]  = (_Float16)((f2.x + g2.x) * scale);
    afrag[kk][9]  = (_Float16)((f2.y + g2.y) * scale);
    afrag[kk][10] = (_Float16)((f2.z + g2.z) * scale);
    afrag[kk][11] = (_Float16)((f2.w + g2.w) * scale);
    afrag[kk][12] = (_Float16)((f3.x + g3.x) * scale);
    afrag[kk][13] = (_Float16)((f3.y + g3.y) * scale);
    afrag[kk][14] = (_Float16)((f3.z + g3.z) * scale);
    afrag[kk][15] = (_Float16)((f3.w + g3.w) * scale);
  }

  // ---- 8 column tiles x 4 WMMA steps (32 wmma / wave) ----
  v8f acc[8];
#pragma unroll
  for (int t = 0; t < 8; ++t) {
    const _Float16* __restrict__ brow = Wh + (long)(t * 16 + lane16) * 128;
    v8f c = {0.f, 0.f, 0.f, 0.f, 0.f, 0.f, 0.f, 0.f};
#pragma unroll
    for (int kk = 0; kk < 4; ++kk) {
      const int kb = kk * 32 + half_sel * 8;
      const v8h bLo = *(const v8h*)(brow + kb);
      const v8h bHi = *(const v8h*)(brow + kb + 16);
      v16h b;
#pragma unroll
      for (int u = 0; u < 8; ++u) { b[u] = bLo[u]; b[u + 8] = bHi[u]; }
      c = __builtin_amdgcn_wmma_f32_16x16x32_f16(
          false, afrag[kk], false, b, (short)0, c, false, false);
    }
    acc[t] = c;
  }

  // ---- bias / gamma / beta for this lane's 8 columns ----
  float biasv[8], gv[8], bv[8];
#pragma unroll
  for (int t = 0; t < 8; ++t) {
    const int col = t * 16 + lane16;
    biasv[t] = bl[col];
    gv[t] = gm[col];
    bv[t] = bt[col];
  }

  const bool fullTile = (rowBase + 16) <= nRows;  // wave-uniform
#pragma unroll
  for (int j = 0; j < 8; ++j) {
    float sum = 0.f, sq = 0.f;
#pragma unroll
    for (int t = 0; t < 8; ++t) {
      const float h = acc[t][j] + biasv[t];
      acc[t][j] = h;
      sum += h;
      sq += h * h;
    }
    // reduce over the 16 lanes that hold row (half_sel*8 + j)
#pragma unroll
    for (int m = 8; m >= 1; m >>= 1) {
      sum += __shfl_xor(sum, m, 32);
      sq  += __shfl_xor(sq,  m, 32);
    }
    const float mean = sum * (1.0f / 128.0f);
    const float var  = sq * (1.0f / 128.0f) - mean * mean;  // biased == jnp.var
    const float rstd = rsqrtf(var + LN_EPS);
    const int row = rowBase + half_sel * 8 + j;
    if (fullTile || row < nRows) {
      float* __restrict__ orow = out + (long)row * 128;
#pragma unroll
      for (int t = 0; t < 8; ++t) {
        float y = (acc[t][j] - mean) * rstd * gv[t] + bv[t];
        y = y > 0.0f ? y : expm1f(y);                // jax.nn.elu, alpha=1
        orow[t * 16 + lane16] = y;
      }
    }
  }
}

// ---------------------------------------------------------------- gather
__global__ void k_gather(const float* __restrict__ feats, const int* __restrict__ uid,
                         float* __restrict__ out, int nU) {
  long tid = (long)blockIdx.x * blockDim.x + threadIdx.x;
  int u = (int)(tid >> 5);
  if (u >= nU) return;
  int lane = (int)(tid & 31);
  const int n = uid[u];
  *(float4*)(out + (long)u * 128 + lane * 4) =
      *(const float4*)(feats + (long)n * 128 + lane * 4);
}

// ---------------------------------------------------------------- launch
extern "C" void kernel_launch(void* const* d_in, const int* in_sizes, int n_in,
                              void* d_out, int out_size, void* d_ws, size_t ws_size,
                              hipStream_t stream) {
  (void)n_in; (void)out_size; (void)ws_size;
  const float* emb   = (const float*)d_in[0];
  const float* W     = (const float*)d_in[1];
  const float* bias  = (const float*)d_in[2];
  const float* gamma = (const float*)d_in[3];
  const float* beta  = (const float*)d_in[4];
  const int*   src   = (const int*)d_in[5];
  const int*   dst   = (const int*)d_in[6];
  const int*   uid   = (const int*)d_in[7];
  float* out = (float*)d_out;

  const int D = 128;
  const int N = in_sizes[0] / D;
  const int E = in_sizes[5];
  const int U = in_sizes[7];
  const int ORDER = in_sizes[1] / (D * D);
  const long NF = (long)N * D;

  // Scratch layout (~51.5 MB): two rotating N x D f32 buffers + inv + f16 W.
  char* ws = (char*)d_ws;
  size_t off = 0;
  auto alignup = [](size_t x) { return (x + 255) & ~(size_t)255; };
  float* bufA = (float*)(ws + off); off = alignup(off + NF * sizeof(float));
  float* bufB = (float*)(ws + off); off = alignup(off + NF * sizeof(float));
  float* inv  = (float*)(ws + off); off = alignup(off + (size_t)N * sizeof(float));
  _Float16* Wh = (_Float16*)(ws + off);
  off = alignup(off + (size_t)ORDER * D * D * sizeof(_Float16));

  const int B = 256;
  auto cdiv = [](long a, long b) { return (int)((a + b - 1) / b); };

  // 1/(in_degree + 1), recomputed each call.
  k_zero_f32<<<cdiv(N, B), B, 0, stream>>>(inv, N);
  k_degree<<<cdiv(E, B), B, 0, stream>>>(dst, inv, E);
  k_inv<<<cdiv(N, B), B, 0, stream>>>(inv, N);

  // Pre-convert all layer weights to f16 (GEMM B operand, rows = out channels).
  k_f32_to_f16<<<cdiv((long)ORDER * D * D, B), B, 0, stream>>>(W, Wh, (long)ORDER * D * D);

  // Layer rotation with out aliased onto the dead neigh buffer:
  //   l0: fin=emb  nb=A out=A | l1: fin=A nb=B out=B | l2: fin=B nb=A out=A
  const int layerWaves = cdiv(N, 16);
  const float* fin = emb;
  float* nbufs[3] = {bufA, bufB, bufA};
  for (int l = 0; l < ORDER; ++l) {
    float* nb = nbufs[l];
    k_zero_f32<<<cdiv(NF, B), B, 0, stream>>>(nb, NF);
    k_scatter<<<cdiv((long)E * 32, B), B, 0, stream>>>(fin, src, dst, nb, E);
    k_sage_layer<<<cdiv((long)layerWaves * 32, B), B, 0, stream>>>(
        fin, nb, inv, Wh + (long)l * D * D, bias + l * D, gamma + l * D,
        beta + l * D, /*out=*/nb, N);
    fin = nb;
  }
  k_gather<<<cdiv((long)U * 32, B), B, 0, stream>>>(fin, uid, out, U);
}
